// ConditionalODE_17927193494457
// MI455X (gfx1250) — matrix operand order repaired and checked by reading, not compiled, not run on hardware
//
// CDNA5 / gfx1250 fused conditional-ODE MLP + exact divergence (2 JVPs).
// Compute-bound (~430 GFLOP GEMM vs ~10MB traffic) -> f16 WMMA w/ f32 accum.
// One persistent workgroup per 16 rows: all activations stay in LDS; weights
// are pre-converted to transposed f16 in d_ws (~1.25 MB) by a prep kernel.
// Round-2 changes: pointer-bump addressing + unroll-1 K loops to keep VGPRs
// <256 (round 1 showed s_set_vgpr_msb thrash from >256 VGPR allocation).
#include <hip/hip_runtime.h>
#include <math.h>

typedef __attribute__((ext_vector_type(16))) _Float16 v16h;
typedef __attribute__((ext_vector_type(8)))  _Float16 v8h;
typedef __attribute__((ext_vector_type(8)))  float    v8f;

#define NROWS   (2048*64)      // B*S
#define HID     512
#define XSTR    552            // xbuf row stride in halves (544 K-pad + 8)
#define USTR    516            // ubuf row stride in floats (bank-conflict pad)
#define KP0     160            // layer-0 K padded (132 -> 160)
#define KP      544            // layers 1-2 K padded (514 -> 544)
#define LN_EPS  1e-5f

#define XBUF_BYTES (48*XSTR*2)            // 52,992
#define UBUF_BYTES (48*USTR*4)            // 99,072
#define SHMEM_BYTES (XBUF_BYTES + UBUF_BYTES + 64)   // 152,128 < 160KB

#define WT0_OFF  0
#define WT1_OFF  (512*KP0*2)              // 163,840
#define WT2_OFF  (WT1_OFF + 512*KP*2)     // 720,896 ; end 1,277,952 bytes of ws

__device__ __forceinline__ v8f wmma16(v16h a, v16h b, v8f c) {
  return __builtin_amdgcn_wmma_f32_16x16x32_f16(false, a, false, b,
                                                (short)0, c, false, false);
}

// 8 contiguous halves at p plus 8 at p+16 halves: matches the documented
// 16-bit A(16x32)/B(32x16) per-lane element->K mapping for wave32 WMMA.
__device__ __forceinline__ v16h load_frag(const _Float16* p) {
  union { v16h v; v8h h[2]; } u;
  u.h[0] = *(const v8h*)(p);
  u.h[1] = *(const v8h*)(p + 16);
  return u.v;
}

__device__ __forceinline__ float wred(float v) {
#pragma unroll
  for (int m = 16; m >= 1; m >>= 1) v += __shfl_xor(v, m, 32);
  return v;
}
__device__ __forceinline__ float softplus_f(float x) {
  return x > 20.f ? x : log1pf(__expf(x));
}
__device__ __forceinline__ float sigmoid_f(float x) {
  return 1.f / (1.f + __expf(-x));
}

// ---- weight prep: f32 row-major [K][512] -> f16 transposed [512][Kpad] ----
__global__ __launch_bounds__(256) void prep_weights(
    const float* __restrict__ W0, const float* __restrict__ W1,
    const float* __restrict__ W2, _Float16* __restrict__ wt0,
    _Float16* __restrict__ wt1, _Float16* __restrict__ wt2) {
  int idx = blockIdx.x * 256 + threadIdx.x;
  if (idx < 512*KP0) {
    int n = idx / KP0, k = idx % KP0;
    wt0[idx] = (k < 132) ? (_Float16)W0[k*512 + n] : (_Float16)0.f;
  }
  if (idx < 512*KP) {
    int n = idx / KP, k = idx % KP;
    wt1[idx] = (k < 514) ? (_Float16)W1[k*512 + n] : (_Float16)0.f;
    wt2[idx] = (k < 514) ? (_Float16)W2[k*512 + n] : (_Float16)0.f;
  }
}

// ---- stacked GEMM: [48 x 544] (primal + 2 tangents) x [544 x 512] ----
// Per wave: 3 M-tiles x 4 N-tiles. Tile offsets are compile-time constants
// folded into load-instruction offset fields; only `k` advances.
__device__ __forceinline__ void gemm_full(const _Float16* __restrict__ wt,
                                          const _Float16* xbuf, float* ubuf,
                                          int wave, int lane) {
  v8f acc[3][4];
  const v8f zero = {0.f,0.f,0.f,0.f,0.f,0.f,0.f,0.f};
#pragma unroll
  for (int mi = 0; mi < 3; ++mi)
#pragma unroll
    for (int ni = 0; ni < 4; ++ni) acc[mi][ni] = zero;

  const int n0   = wave * 64;
  const int arow = lane & 15;
  const _Float16* abase = xbuf + arow*XSTR + ((lane >> 4) << 3);
  const _Float16* bbase = wt + (size_t)(n0 + arow)*KP + ((lane >> 4) << 3);

#pragma unroll 1
  for (int k = 0; k < KP; k += 32) {
    v16h a0 = load_frag(abase + k);
    v16h a1 = load_frag(abase + 16*XSTR + k);
    v16h a2 = load_frag(abase + 32*XSTR + k);
#pragma unroll
    for (int ni = 0; ni < 4; ++ni) {
      v16h b = load_frag(bbase + ni*16*KP + k);
      acc[0][ni] = wmma16(a0, b, acc[0][ni]);
      acc[1][ni] = wmma16(a1, b, acc[1][ni]);
      acc[2][ni] = wmma16(a2, b, acc[2][ni]);
    }
  }

  const int rbase = (lane >> 4) << 3;    // C/D layout: lanes>=16 hold M+8
  float* ub = ubuf + n0 + arow;
#pragma unroll
  for (int mi = 0; mi < 3; ++mi)
#pragma unroll
    for (int ni = 0; ni < 4; ++ni)
#pragma unroll
      for (int r = 0; r < 8; ++r)
        ub[(mi*16 + rbase + r)*USTR + ni*16] = acc[mi][ni][r];
}

__global__ __launch_bounds__(256, 2) void ode_fused(
    const float* __restrict__ t_p, const float* __restrict__ z,
    const float* __restrict__ cond, const float* __restrict__ W0,
    const float* __restrict__ b0, const float* __restrict__ g0, const float* __restrict__ be0,
    const float* __restrict__ b1, const float* __restrict__ g1, const float* __restrict__ be1,
    const float* __restrict__ b2, const float* __restrict__ g2, const float* __restrict__ be2,
    const float* __restrict__ W3, const float* __restrict__ b3,
    const _Float16* __restrict__ wt0, const _Float16* __restrict__ wt1,
    const _Float16* __restrict__ wt2,
    float* __restrict__ out, float* __restrict__ odiv)
{
  extern __shared__ char smem[];
  _Float16* xbuf = (_Float16*)smem;                       // [48][XSTR] f16
  float*    ubuf = (float*)(smem + XBUF_BYTES);           // [48][USTR] f32
  float*    rsbuf = (float*)(smem + XBUF_BYTES + UBUF_BYTES);  // [16]

  const int tid = threadIdx.x;
  const int wave = tid >> 5, lane = tid & 31;
  const int wg = blockIdx.x;
  const float te = t_p[0];

  // zero xbuf once: provides K-padding zeros + zero te/pos cols for tangents
  {
    unsigned int* p = (unsigned int*)xbuf;
    for (int i = tid; i < XBUF_BYTES/4; i += 256) p[i] = 0u;
  }
  __syncthreads();

  // fill layer-0 primal rows: [te, pos, z0, z1, cond(128)] -> f16
#pragma unroll
  for (int rr = 0; rr < 2; ++rr) {
    int r = wave*2 + rr;
    int grow = wg*16 + r;
    float pos = (float)((grow & 63) + 1) * (1.f/64.f);
    for (int c = lane; c < 132; c += 32) {
      float v;
      if (c == 0)      v = te;
      else if (c == 1) v = pos;
      else if (c < 4)  v = z[grow*2 + (c-2)];
      else             v = cond[(grow >> 6)*128 + (c-4)];
      xbuf[r*XSTR + c] = (_Float16)v;
    }
  }
  __syncthreads();

  // layer-0 GEMM (primal only): [16 x 160] x [160 x 512]
  {
    v8f acc[4];
    const v8f zero = {0.f,0.f,0.f,0.f,0.f,0.f,0.f,0.f};
#pragma unroll
    for (int ni = 0; ni < 4; ++ni) acc[ni] = zero;
    const int n0   = wave * 64;
    const int arow = lane & 15;
    const _Float16* abase = xbuf + arow*XSTR + ((lane >> 4) << 3);
    const _Float16* bbase = wt0 + (size_t)(n0 + arow)*KP0 + ((lane >> 4) << 3);
#pragma unroll 1
    for (int k = 0; k < KP0; k += 32) {
      v16h a = load_frag(abase + k);
#pragma unroll
      for (int ni = 0; ni < 4; ++ni) {
        v16h b = load_frag(bbase + ni*16*KP0 + k);
        acc[ni] = wmma16(a, b, acc[ni]);
      }
    }
    const int rbase = (lane >> 4) << 3;
    float* ub = ubuf + n0 + arow;
#pragma unroll
    for (int ni = 0; ni < 4; ++ni)
#pragma unroll
      for (int r = 0; r < 8; ++r)
        ub[(rbase + r)*USTR + ni*16] = acc[ni][r];
  }
  __syncthreads();

  const float* bias[3] = {b0, b1, b2};
  const float* gam[3]  = {g0, g1, g2};
  const float* bet[3]  = {be0, be1, be2};

  for (int layer = 0; layer < 3; ++layer) {
    const float* bi = bias[layer];
    const float* g  = gam[layer];
    const float* be = bet[layer];

    // phase A: primal rows (2 per wave): bias + LN stats + softplus.
    // stores nrm in-place into ubuf; y as f16 into xbuf (layers 0,1).
#pragma unroll
    for (int rr = 0; rr < 2; ++rr) {
      int r = wave*2 + rr;
      float uv[16]; float s = 0.f, s2 = 0.f;
#pragma unroll
      for (int i = 0; i < 16; ++i) {
        int c = lane + (i << 5);
        float u = ubuf[r*USTR + c] + bi[c];
        uv[i] = u; s += u; s2 += u*u;
      }
      s = wred(s); s2 = wred(s2);
      float mu  = s  * (1.f/512.f);
      float var = s2 * (1.f/512.f) - mu*mu;
      float rs  = rsqrtf(var + LN_EPS);
      if (lane == 0) rsbuf[r] = rs;
#pragma unroll
      for (int i = 0; i < 16; ++i) {
        int c = lane + (i << 5);
        float nrm = (uv[i] - mu) * rs;
        ubuf[r*USTR + c] = nrm;                       // keep for JVP / final
        if (layer < 2) {
          float L = nrm * g[c] + be[c];
          xbuf[r*XSTR + 2 + c] = (_Float16)softplus_f(L);
        }
      }
    }
    __syncthreads();

    // phase B: tangent rows (4 per wave): LN-JVP + softplus-JVP.
    // layer 0 tangent pre-activation is W0 row (2+d) (analytic, no GEMM).
#pragma unroll
    for (int j = 0; j < 4; ++j) {
      int tr = 16 + wave*4 + j;
      int p  = tr & 15;
      int d  = (tr >> 4) - 1;
      float duv[16], nv[16]; float sa = 0.f, sb = 0.f;
#pragma unroll
      for (int i = 0; i < 16; ++i) {
        int c = lane + (i << 5);
        float du  = (layer == 0) ? W0[(2 + d)*HID + c] : ubuf[tr*USTR + c];
        float nrm = ubuf[p*USTR + c];
        duv[i] = du; nv[i] = nrm;
        sa += du; sb += nrm*du;
      }
      sa = wred(sa); sb = wred(sb);
      float am = sa * (1.f/512.f), bm = sb * (1.f/512.f);
      float rs = rsbuf[p];
#pragma unroll
      for (int i = 0; i < 16; ++i) {
        int c = lane + (i << 5);
        float nrm = nv[i];
        float L  = nrm * g[c] + be[c];
        float dL = g[c] * rs * (duv[i] - am - nrm*bm);
        float dy = sigmoid_f(L) * dL;
        if (layer < 2) xbuf[tr*XSTR + 2 + c] = (_Float16)dy;
        else           ubuf[tr*USTR + c] = dy;        // dy3 kept f32 for L3
      }
    }
    __syncthreads();

    if (layer < 2) {
      gemm_full((layer == 0) ? wt1 : wt2, xbuf, ubuf, wave, lane);
      __syncthreads();
    }
  }

  // layer 3 (512 -> 2) + divergence: VALU dots, f32 precision.
#pragma unroll
  for (int rr = 0; rr < 2; ++rr) {
    int r = wave*2 + rr;
    float s0 = 0.f, s1 = 0.f, dv0 = 0.f, dv1 = 0.f;
#pragma unroll
    for (int i = 0; i < 16; ++i) {
      int c = lane + (i << 5);
      float nrm = ubuf[r*USTR + c];
      float y = softplus_f(nrm * g2[c] + be2[c]);     // recompute y3 in f32
      float w30 = W3[(2 + c)*2 + 0];
      float w31 = W3[(2 + c)*2 + 1];
      s0 += y*w30; s1 += y*w31;
      dv0 += ubuf[(16 + r)*USTR + c] * w30;           // d/dz0 -> out comp 0
      dv1 += ubuf[(32 + r)*USTR + c] * w31;           // d/dz1 -> out comp 1
    }
    s0 = wred(s0); s1 = wred(s1); dv0 = wred(dv0); dv1 = wred(dv1);
    if (lane == 0) {
      int grow = wg*16 + r;
      float pos = (float)((grow & 63) + 1) * (1.f/64.f);
      out[grow*2 + 0] = te*W3[0] + pos*W3[2] + s0 + b3[0];
      out[grow*2 + 1] = te*W3[1] + pos*W3[3] + s1 + b3[1];
      odiv[grow] = -(dv0 + dv1);
    }
  }
}

extern "C" void kernel_launch(void* const* d_in, const int* in_sizes, int n_in,
                              void* d_out, int out_size, void* d_ws, size_t ws_size,
                              hipStream_t stream) {
  (void)in_sizes; (void)n_in; (void)out_size; (void)ws_size;
  const float* t    = (const float*)d_in[0];
  const float* z    = (const float*)d_in[1];
  const float* cond = (const float*)d_in[2];
  const float* W0   = (const float*)d_in[3];
  const float* b0   = (const float*)d_in[4];
  const float* g0   = (const float*)d_in[5];
  const float* be0  = (const float*)d_in[6];
  const float* W1   = (const float*)d_in[7];
  const float* b1   = (const float*)d_in[8];
  const float* g1   = (const float*)d_in[9];
  const float* be1  = (const float*)d_in[10];
  const float* W2   = (const float*)d_in[11];
  const float* b2   = (const float*)d_in[12];
  const float* g2   = (const float*)d_in[13];
  const float* be2  = (const float*)d_in[14];
  const float* W3   = (const float*)d_in[15];
  const float* b3   = (const float*)d_in[16];

  _Float16* wt0 = (_Float16*)((char*)d_ws + WT0_OFF);
  _Float16* wt1 = (_Float16*)((char*)d_ws + WT1_OFF);
  _Float16* wt2 = (_Float16*)((char*)d_ws + WT2_OFF);

  float* out  = (float*)d_out;
  float* odiv = out + (size_t)NROWS*2;

  prep_weights<<<(512*KP + 255)/256, 256, 0, stream>>>(W0, W1, W2, wt0, wt1, wt2);

  (void)hipFuncSetAttribute((const void*)ode_fused,
                            hipFuncAttributeMaxDynamicSharedMemorySize,
                            SHMEM_BYTES);
  ode_fused<<<NROWS/16, 256, SHMEM_BYTES, stream>>>(
      t, z, cond, W0, b0, g0, be0, b1, g1, be1, b2, g2, be2, W3, b3,
      wt0, wt1, wt2, out, odiv);
}
